// PointPillarsEncoder_3582002725065
// MI455X (gfx1250) — matrix operand order repaired and checked by reading, not compiled, not run on hardware
//
#include <hip/hip_runtime.h>
#include <hip/hip_bf16.h>

// ---------------------------------------------------------------------------
// PointPillars encoder for MI455X (gfx1250, wave32, WMMA).
// All dense math runs through v_wmma_f32_16x16x32_f16 (f16 in, f32 acc).
// GEMM uses 32x64 register macro-tiles (8 WMMA per k-step, ~21 FLOP/B of
// fragment traffic) with __launch_bounds__(256,1) so the 64 accumulator +
// ~96 fragment/pipeline VGPRs fit without scratch spills. Attention is
// processed per (batch, head) so the 64MB score slice stays in the 192MB L2.
// PointNet chunked by 1024 pillars to stay L2-resident vs 23.3 TB/s HBM.
// ---------------------------------------------------------------------------

typedef __attribute__((ext_vector_type(16))) _Float16     v16h;
typedef __attribute__((ext_vector_type(8)))  float        v8f;
typedef __attribute__((ext_vector_type(4)))  unsigned int v4u;

#define VOXEL_INV 5.0f     // 1 / 0.2
#define GDIM      64
#define PCAP      100
#define BATCH     2
#define DMODEL    64
#define NHEAD     4
#define DHEAD     16
#define DHEAD_PAD 32       // pad K to one WMMA k-step
#define SEQ       (GDIM * GDIM)          // 4096
#define FFDIM     2048
#define NPILLARS  (BATCH * GDIM * GDIM)  // 8192
#define CHUNK_PIL 1024
#define CHUNK_SLOTS (CHUNK_PIL * PCAP)   // 102400 (multiple of 32)

union FragH { v16h v; v4u u[2]; _Float16 h[16]; };
union FragF { v8f  v; float f[8]; };

// ---- WMMA fragment loaders -------------------------------------------------
// A: row-major [M, lda] f16. 16x32 fragment at (row0, k0).
// lane<16: row=lane, e0..7 -> K=k0..k0+7, e8..15 -> K=k0+16..k0+23
// lane>=16: row=lane-16, K ranges shifted by +8.
__device__ __forceinline__ v16h load_frag_a(const _Float16* __restrict__ A,
                                            int lda, int row0, int k0) {
  int lane = threadIdx.x & 31;
  const _Float16* p = A + (size_t)(row0 + (lane & 15)) * lda
                        + (k0 + ((lane >> 4) << 3));
  FragH f;
  f.u[0] = *(const v4u*)(p);        // 8 halves, 16B
  f.u[1] = *(const v4u*)(p + 16);   // 8 halves, 16B
  return f.v;
}

// B: weights kept as [N, K] row-major (i.e. W[dout, din] == B^T), ldb = K.
// lane: N = n0 + lane%16; e0..7 -> K=k0+8*(lane/16)+0..7, e8..15 -> +16.
__device__ __forceinline__ v16h load_frag_b(const _Float16* __restrict__ Bw,
                                            int ldb, int n0, int k0) {
  int lane = threadIdx.x & 31;
  const _Float16* p = Bw + (size_t)(n0 + (lane & 15)) * ldb
                         + (k0 + ((lane >> 4) << 3));
  FragH f;
  f.u[0] = *(const v4u*)(p);        // halves 0..7
  f.u[1] = *(const v4u*)(p + 16);   // halves 8..15
  return f.v;
}

// ---- Register-blocked WMMA GEMM: C[M,N] = act(A[M,K] @ Bw[N,K]^T + bias) --
// One wave computes a (16*MT) x (16*NT) macro-tile: MT A-frags + NT B-frags
// loaded per k-step, MT*NT WMMAs issued. Tile selection is wave-uniform so
// EXEC is all-ones around every v_wmma (ISA requirement).
// __launch_bounds__(256, 1): min 1 wave/EU -> full VGPR budget, no spills;
// latency is hidden by clause-pipelined b128 loads, not wave count.
template <int MT, int NT>
__global__ __launch_bounds__(256, 1)
void gemm_wmma_kernel(const _Float16* __restrict__ A, int lda,
                      const _Float16* __restrict__ Bw, int ldb,
                      const float* __restrict__ bias,
                      float* __restrict__ o32,
                      _Float16* __restrict__ o16,
                      int ldc, int M, int N, int K, int relu) {
  int gwave  = (int)((blockIdx.x * blockDim.x + threadIdx.x) >> 5);
  int tilesN = N / (16 * NT);
  int tilesM = M / (16 * MT);
  if (gwave >= tilesM * tilesN) return;       // wave-uniform exit
  int tm = gwave / tilesN;
  int tn = gwave - tm * tilesN;
  int row0 = tm * (16 * MT), n0 = tn * (16 * NT);

  v8f zero = {};
  v8f acc[MT][NT];
#pragma unroll
  for (int i = 0; i < MT; ++i)
#pragma unroll
    for (int j = 0; j < NT; ++j) acc[i][j] = zero;

  for (int k0 = 0; k0 < K; k0 += 32) {
    v16h a[MT];
    v16h b[NT];
#pragma unroll
    for (int i = 0; i < MT; ++i) a[i] = load_frag_a(A, lda, row0 + 16 * i, k0);
#pragma unroll
    for (int j = 0; j < NT; ++j) b[j] = load_frag_b(Bw, ldb, n0 + 16 * j, k0);
#pragma unroll
    for (int i = 0; i < MT; ++i)
#pragma unroll
      for (int j = 0; j < NT; ++j)
        acc[i][j] = __builtin_amdgcn_wmma_f32_16x16x32_f16(
            /*neg_a=*/false, a[i], /*neg_b=*/false, b[j],
            /*c_mod=*/(short)0, acc[i][j],
            /*reuse_a=*/false, /*reuse_b=*/false);
  }

  // C layout: VGPR r -> M = row0 + 8*(lane/16) + r, N = n0 + lane%16
  int lane = threadIdx.x & 31;
#pragma unroll
  for (int i = 0; i < MT; ++i) {
#pragma unroll
    for (int j = 0; j < NT; ++j) {
      int n  = n0 + 16 * j + (lane & 15);
      int m0 = row0 + 16 * i + ((lane >> 4) << 3);
      float bv = bias ? bias[n] : 0.0f;
      FragF cf; cf.v = acc[i][j];
#pragma unroll
      for (int r = 0; r < 8; ++r) {
        float v = cf.f[r] + bv;
        if (relu) v = fmaxf(v, 0.0f);
        size_t o = (size_t)(m0 + r) * ldc + n;
        if (o32) o32[o] = v;
        if (o16) o16[o] = (_Float16)v;
      }
    }
  }
}

// ---- f32 -> f16 weight convert --------------------------------------------
__global__ void f32_to_f16_kernel(const float* __restrict__ s,
                                  _Float16* __restrict__ d, int n) {
  int i = blockIdx.x * blockDim.x + threadIdx.x;
  if (i < n) d[i] = (_Float16)s[i];
}

// ---- Voxelization: point -> pillar scatter via atomic slot counters -------
__global__ void voxelize_kernel(const float* __restrict__ pc,
                                const int* __restrict__ boff,
                                int* __restrict__ counts,
                                float* __restrict__ vox, int n) {
  int i = blockIdx.x * blockDim.x + threadIdx.x;
  if (i >= n) return;
  float x = pc[i * 4 + 0], y = pc[i * 4 + 1];
  float z = pc[i * 4 + 2], w = pc[i * 4 + 3];
  int ix = (int)(x * VOXEL_INV); ix = min(max(ix, 0), GDIM - 1);
  int iy = (int)(y * VOXEL_INV); iy = min(max(iy, 0), GDIM - 1);
  int flat = boff[i] * (GDIM * GDIM) + ix * GDIM + iy;
  int slot = atomicAdd(&counts[flat], 1);
  if (slot < PCAP) {
    float* d = vox + ((size_t)flat * PCAP + slot) * 4;
    d[0] = x; d[1] = y; d[2] = z; d[3] = w;
  }
}

// ---- fc1: 4 -> 64 (K too small for WMMA; VALU) ----------------------------
__global__ void fc1_kernel(const float* __restrict__ vox,
                           const float* __restrict__ W,
                           const float* __restrict__ bias,
                           _Float16* __restrict__ out, int rows) {
  int i = blockIdx.x * blockDim.x + threadIdx.x;
  if (i >= rows) return;
  float x0 = vox[i * 4 + 0], x1 = vox[i * 4 + 1];
  float x2 = vox[i * 4 + 2], x3 = vox[i * 4 + 3];
#pragma unroll 4
  for (int j = 0; j < 64; ++j) {
    float a = bias[j] + W[j * 4 + 0] * x0 + W[j * 4 + 1] * x1
                      + W[j * 4 + 2] * x2 + W[j * 4 + 3] * x3;
    out[(size_t)i * 64 + j] = (_Float16)fmaxf(a, 0.0f);
  }
}

// ---- masked max-pool over P slots -> pillar features (f32 + f16) ----------
__global__ void maxpool_kernel(const _Float16* __restrict__ h3,
                               const int* __restrict__ counts,
                               float* __restrict__ x32,
                               _Float16* __restrict__ x16, int pillar0) {
  int idx = blockIdx.x * blockDim.x + threadIdx.x;
  if (idx >= CHUNK_PIL * DMODEL) return;
  int d  = idx & (DMODEL - 1);
  int lp = idx >> 6;
  int pg = pillar0 + lp;
  float m = -3.0e38f;
  for (int p = 0; p < PCAP; ++p)
    m = fmaxf(m, (float)h3[((size_t)lp * PCAP + p) * DMODEL + d]);
  if (counts[pg] == 0) m = 0.0f;   // reference: whole-pillar mask
  x32[(size_t)pg * DMODEL + d] = m;
  x16[(size_t)pg * DMODEL + d] = (_Float16)m;
}

// ---- qkv [B*S,192] -> Qpad/Kpad [B,H,S,32] (zero pad), Vt [B,H,16,S] ------
__global__ void qkv_split_kernel(const _Float16* __restrict__ qkv,
                                 _Float16* __restrict__ Qp,
                                 _Float16* __restrict__ Kp,
                                 _Float16* __restrict__ Vt) {
  int idx = blockIdx.x * blockDim.x + threadIdx.x;   // over B*S*NH*DHEAD
  if (idx >= BATCH * SEQ * NHEAD * DHEAD) return;
  int d  = idx & (DHEAD - 1);
  int h  = (idx >> 4) & (NHEAD - 1);
  int bs = idx >> 6;                 // b*SEQ + s
  int b  = bs >> 12;                 // SEQ == 4096
  int s  = bs & (SEQ - 1);
  const _Float16* src = qkv + (size_t)bs * (3 * DMODEL);
  size_t qb = (((size_t)(b * NHEAD + h) * SEQ) + s) * DHEAD_PAD;
  Qp[qb + d]         = src[h * DHEAD + d];
  Qp[qb + DHEAD + d] = (_Float16)0.0f;
  Kp[qb + d]         = src[DMODEL + h * DHEAD + d];
  Kp[qb + DHEAD + d] = (_Float16)0.0f;
  Vt[((size_t)(b * NHEAD + h) * DHEAD + d) * SEQ + s] =
      src[2 * DMODEL + h * DHEAD + d];
}

// ---- row softmax: scores f32 [S] -> probs f16, with 1/sqrt(dh) scale ------
__global__ void softmax_kernel(const float* __restrict__ scores,
                               _Float16* __restrict__ probs, float scale) {
  __shared__ float red[256];
  int t = threadIdx.x;
  const float* row = scores + (size_t)blockIdx.x * SEQ;
  _Float16* prow   = probs  + (size_t)blockIdx.x * SEQ;

  float mx = -3.0e38f;
  for (int i = t; i < SEQ; i += 256) mx = fmaxf(mx, row[i] * scale);
  red[t] = mx; __syncthreads();
  for (int s = 128; s > 0; s >>= 1) {
    if (t < s) red[t] = fmaxf(red[t], red[t + s]);
    __syncthreads();
  }
  float rmax = red[0]; __syncthreads();

  float sum = 0.0f;
  for (int i = t; i < SEQ; i += 256) sum += __expf(row[i] * scale - rmax);
  red[t] = sum; __syncthreads();
  for (int s = 128; s > 0; s >>= 1) {
    if (t < s) red[t] += red[t + s];
    __syncthreads();
  }
  float inv = 1.0f / red[0];

  for (int i = t; i < SEQ; i += 256)
    prow[i] = (_Float16)(__expf(row[i] * scale - rmax) * inv);
}

// ---- LayerNorm(x + f) -> xo (f32) and xo16 (f16), D=64, one block/row -----
__global__ void ln_kernel(const float* __restrict__ x,
                          const float* __restrict__ f,
                          const float* __restrict__ g,
                          const float* __restrict__ b,
                          float* __restrict__ xo,
                          _Float16* __restrict__ xo16) {
  __shared__ float s1[64];
  __shared__ float s2[64];
  int t = threadIdx.x;
  size_t base = (size_t)blockIdx.x * DMODEL;
  float v = x[base + t] + f[base + t];
  s1[t] = v; s2[t] = v * v; __syncthreads();
  for (int s = 32; s > 0; s >>= 1) {
    if (t < s) { s1[t] += s1[t + s]; s2[t] += s2[t + s]; }
    __syncthreads();
  }
  float mean = s1[0] * (1.0f / DMODEL);
  float var  = s2[0] * (1.0f / DMODEL) - mean * mean;
  float y = (v - mean) * rsqrtf(var + 1e-5f) * g[t] + b[t];
  xo[base + t]   = y;
  xo16[base + t] = (_Float16)y;
}

// ---------------------------------------------------------------------------
// Host launcher
// ---------------------------------------------------------------------------
// Input flattening (setup_inputs dict insertion order):
//  0 point_clouds, 1 batch_offsets,
//  2 fc1.w, 3 fc1.b, 4 fc2.w, 5 fc2.b, 6 fc3.w, 7 fc3.b,
//  then per layer (stride 12 starting at 8):
//  qkv.w, qkv.b, out.w, out.b, ln1_g, ln1_b, ff1.w, ff1.b, ff2.w, ff2.b,
//  ln2_g, ln2_b
extern "C" void kernel_launch(void* const* d_in, const int* in_sizes, int n_in,
                              void* d_out, int out_size, void* d_ws,
                              size_t ws_size, hipStream_t stream) {
  (void)n_in; (void)out_size; (void)ws_size;

  const float* pc   = (const float*)d_in[0];
  const int*   boff = (const int*)d_in[1];
  const float* fc1w = (const float*)d_in[2];
  const float* fc1b = (const float*)d_in[3];
  const float* fc2w = (const float*)d_in[4];
  const float* fc2b = (const float*)d_in[5];
  const float* fc3w = (const float*)d_in[6];
  const float* fc3b = (const float*)d_in[7];

  // ---- workspace bump allocator (256B aligned) ----
  char*  wsb = (char*)d_ws;
  size_t off = 0;
  auto alloc = [&](size_t bytes) -> void* {
    off = (off + 255) & ~(size_t)255;
    void* p = wsb + off;
    off += bytes;
    return p;
  };

  int*      counts = (int*)alloc(NPILLARS * sizeof(int));
  float*    vox    = (float*)alloc((size_t)NPILLARS * PCAP * 4 * sizeof(float));
  _Float16* h1     = (_Float16*)alloc((size_t)CHUNK_SLOTS * 64 * 2);
  _Float16* h2     = (_Float16*)alloc((size_t)CHUNK_SLOTS * 128 * 2);
  _Float16* h3     = (_Float16*)alloc((size_t)CHUNK_SLOTS * 64 * 2);
  float*    x32    = (float*)alloc((size_t)NPILLARS * DMODEL * 4);
  _Float16* x16    = (_Float16*)alloc((size_t)NPILLARS * DMODEL * 2);
  _Float16* qkv16  = (_Float16*)alloc((size_t)NPILLARS * 3 * DMODEL * 2);
  _Float16* Qp     = (_Float16*)alloc((size_t)BATCH * NHEAD * SEQ * DHEAD_PAD * 2);
  _Float16* Kp     = (_Float16*)alloc((size_t)BATCH * NHEAD * SEQ * DHEAD_PAD * 2);
  _Float16* Vt     = (_Float16*)alloc((size_t)BATCH * NHEAD * DHEAD * SEQ * 2);
  float*    scores = (float*)alloc((size_t)SEQ * SEQ * 4);
  _Float16* probs  = (_Float16*)alloc((size_t)SEQ * SEQ * 2);
  _Float16* o16    = (_Float16*)alloc((size_t)NPILLARS * DMODEL * 2);
  float*    proj   = (float*)alloc((size_t)NPILLARS * DMODEL * 4);
  _Float16* hff    = (_Float16*)alloc((size_t)NPILLARS * FFDIM * 2);
  float*    ffo    = (float*)alloc((size_t)NPILLARS * DMODEL * 4);
  _Float16* w2h    = (_Float16*)alloc((size_t)128 * 64 * 2);
  _Float16* w3h    = (_Float16*)alloc((size_t)64 * 128 * 2);
  _Float16* wqkvh[2]; _Float16* wouth[2]; _Float16* wff1h[2]; _Float16* wff2h[2];
  for (int l = 0; l < 2; ++l) {
    wqkvh[l] = (_Float16*)alloc((size_t)192 * 64 * 2);
    wouth[l] = (_Float16*)alloc((size_t)64 * 64 * 2);
    wff1h[l] = (_Float16*)alloc((size_t)FFDIM * 64 * 2);
    wff2h[l] = (_Float16*)alloc((size_t)64 * FFDIM * 2);
  }

  const int npts = in_sizes[0] / 4;

  auto cvt = [&](const float* s, _Float16* d, int n) {
    f32_to_f16_kernel<<<(n + 255) / 256, 256, 0, stream>>>(s, d, n);
  };
  // dispatch: 32x64 macro-tile when shapes allow, 32x16 for narrow-N (attn O)
  auto gemm = [&](const _Float16* A, int lda, const _Float16* Bw, int ldb,
                  const float* bias, float* o32p, _Float16* o16p, int ldc,
                  int M, int N, int K, int relu) {
    if (M % 32 == 0 && N % 64 == 0) {
      int tiles  = (M / 32) * (N / 64);
      int blocks = (tiles + 7) / 8;         // 8 waves (256 threads) per block
      gemm_wmma_kernel<2, 4><<<blocks, 256, 0, stream>>>(
          A, lda, Bw, ldb, bias, o32p, o16p, ldc, M, N, K, relu);
    } else if (M % 32 == 0 && N % 16 == 0) {
      int tiles  = (M / 32) * (N / 16);
      int blocks = (tiles + 7) / 8;
      gemm_wmma_kernel<2, 1><<<blocks, 256, 0, stream>>>(
          A, lda, Bw, ldb, bias, o32p, o16p, ldc, M, N, K, relu);
    } else {
      int tiles  = (M / 16) * (N / 16);
      int blocks = (tiles + 7) / 8;
      gemm_wmma_kernel<1, 1><<<blocks, 256, 0, stream>>>(
          A, lda, Bw, ldb, bias, o32p, o16p, ldc, M, N, K, relu);
    }
  };

  // ---- weight conversion (f32 -> f16), recomputed every call ----
  cvt(fc2w, w2h, 128 * 64);
  cvt(fc3w, w3h, 64 * 128);
  for (int l = 0; l < 2; ++l) {
    int base = 8 + 12 * l;
    cvt((const float*)d_in[base + 0], wqkvh[l], 192 * 64);
    cvt((const float*)d_in[base + 2], wouth[l], 64 * 64);
    cvt((const float*)d_in[base + 6], wff1h[l], FFDIM * 64);
    cvt((const float*)d_in[base + 8], wff2h[l], 64 * FFDIM);
  }

  // ---- voxelization ----
  hipMemsetAsync(counts, 0, NPILLARS * sizeof(int), stream);
  hipMemsetAsync(vox, 0, (size_t)NPILLARS * PCAP * 4 * sizeof(float), stream);
  voxelize_kernel<<<(npts + 255) / 256, 256, 0, stream>>>(pc, boff, counts,
                                                          vox, npts);

  // ---- PointNet (chunked by 1024 pillars to stay L2-resident) ----
  for (int c = 0; c < NPILLARS / CHUNK_PIL; ++c) {
    int slot0 = c * CHUNK_SLOTS;
    fc1_kernel<<<(CHUNK_SLOTS + 255) / 256, 256, 0, stream>>>(
        vox + (size_t)slot0 * 4, fc1w, fc1b, h1, CHUNK_SLOTS);
    gemm(h1, 64, w2h, 64, fc2b, nullptr, h2, 128, CHUNK_SLOTS, 128, 64, 1);
    gemm(h2, 128, w3h, 128, fc3b, nullptr, h3, 64, CHUNK_SLOTS, 64, 128, 0);
    maxpool_kernel<<<(CHUNK_PIL * DMODEL + 255) / 256, 256, 0, stream>>>(
        h3, counts, x32, x16, c * CHUNK_PIL);
  }

  // ---- transformer encoder (2 layers, post-norm) ----
  const float inv_sqrt_dh = 0.25f;   // 1/sqrt(16)
  for (int l = 0; l < 2; ++l) {
    int base = 8 + 12 * l;
    const float* bqkv = (const float*)d_in[base + 1];
    const float* bout = (const float*)d_in[base + 3];
    const float* ln1g = (const float*)d_in[base + 4];
    const float* ln1b = (const float*)d_in[base + 5];
    const float* bff1 = (const float*)d_in[base + 7];
    const float* bff2 = (const float*)d_in[base + 9];
    const float* ln2g = (const float*)d_in[base + 10];
    const float* ln2b = (const float*)d_in[base + 11];

    // QKV projection + head split (with DH padded to one WMMA k-step)
    gemm(x16, 64, wqkvh[l], 64, bqkv, nullptr, qkv16, 192,
         NPILLARS, 192, 64, 0);
    qkv_split_kernel<<<(BATCH * SEQ * NHEAD * DHEAD + 255) / 256, 256, 0,
                       stream>>>(qkv16, Qp, Kp, Vt);

    // attention, one (batch, head) slice at a time (64MB scores, L2-resident)
    for (int bh = 0; bh < BATCH * NHEAD; ++bh) {
      int b = bh / NHEAD, h = bh % NHEAD;
      const _Float16* Qbh = Qp + (size_t)bh * SEQ * DHEAD_PAD;
      const _Float16* Kbh = Kp + (size_t)bh * SEQ * DHEAD_PAD;
      const _Float16* Vbh = Vt + (size_t)bh * DHEAD * SEQ;
      _Float16* Obh = o16 + (size_t)b * SEQ * DMODEL + h * DHEAD;
      // scores = Q @ K^T  (K stored [S, 32] row-major == B[N,K])
      gemm(Qbh, DHEAD_PAD, Kbh, DHEAD_PAD, nullptr, scores, nullptr, SEQ,
           SEQ, SEQ, DHEAD_PAD, 0);
      softmax_kernel<<<SEQ, 256, 0, stream>>>(scores, probs, inv_sqrt_dh);
      // O = probs @ V  (Vt is [DH, S] == B[N=16, K=S]) -> 32x16 macro-tile
      gemm(probs, SEQ, Vbh, SEQ, nullptr, nullptr, Obh, DMODEL,
           SEQ, DHEAD, SEQ, 0);
    }

    // out-projection, residual + LN1
    gemm(o16, 64, wouth[l], 64, bout, proj, nullptr, 64,
         NPILLARS, 64, 64, 0);
    ln_kernel<<<NPILLARS, 64, 0, stream>>>(x32, proj, ln1g, ln1b, x32, x16);

    // FFN: relu(x @ W1^T + b1) @ W2^T + b2, residual + LN2
    gemm(x16, 64, wff1h[l], 64, bff1, nullptr, hff, FFDIM,
         NPILLARS, FFDIM, 64, 1);
    gemm(hff, FFDIM, wff2h[l], FFDIM, bff2, ffo, nullptr, 64,
         NPILLARS, 64, FFDIM, 0);
    float* xdst = (l == 1) ? (float*)d_out : x32;   // final LN writes output
    ln_kernel<<<NPILLARS, 64, 0, stream>>>(x32, ffo, ln2g, ln2b, xdst, x16);
  }
}
